// LocalAttention_16904991277195
// MI455X (gfx1250) — matrix operand order repaired and checked by reading, not compiled, not run on hardware
//
#include <hip/hip_runtime.h>
#include <hip/hip_bf16.h>

// LocalAttention for MI455X (gfx1250, wave32, WMMA 16x16x32 f16).
//
// Pipeline (all on `stream`):
//   1) f32->f16 convert: x, Wq, Wk, Wv, Wo into workspace
//   2) zero-fill the attn region of d_out (float4 stores)
//   3) fused QKV GEMM: A-strip staged via TENSOR_LOAD_TO_LDS (TDM), WMMA,
//      bias + RoPE epilogue (1/sqrt(HD) folded into q), per-head f16 out
//   4) sliding-window attention: K tiles staged per-wave with
//      GLOBAL_LOAD_ASYNC_TO_LDS_B128, scores via WMMA, exact softmax in LDS,
//      windowed attn to HBM, P*V via WMMA from LDS
//   5) O-projection GEMM + bias -> fp32 output
//
// Workspace usage: ~48 MB.

#define B_  2
#define S_  2048
#define D_  1024
#define H_  16
#define HD_ 64
#define WIN_ 256

typedef __attribute__((ext_vector_type(16))) _Float16 v16h;
typedef __attribute__((ext_vector_type(8)))  _Float16 v8h;
typedef __attribute__((ext_vector_type(8)))  float    v8f;
typedef unsigned int v4u __attribute__((ext_vector_type(4)));
typedef int          v8i __attribute__((ext_vector_type(8)));
typedef int          v4i __attribute__((ext_vector_type(4)));

// ---------------------------------------------------------------------------
// Fragment helpers (wave32 WMMA register layouts, CDNA5 ISA 7.12.2)
// ---------------------------------------------------------------------------
__device__ __forceinline__ v16h load_a_frag(const _Float16* base, int ld, int lane) {
    const int m  = lane & 15;
    const int kb = (lane < 16) ? 0 : 8;
    const _Float16* p = base + m * ld + kb;
    v8h lo = *(const v8h*)(p);
    v8h hi = *(const v8h*)(p + 16);
    return __builtin_shufflevector(lo, hi, 0, 1, 2, 3, 4, 5, 6, 7,
                                           8, 9, 10, 11, 12, 13, 14, 15);
}

__device__ __forceinline__ v16h load_b_frag(const _Float16* base, int ld, int lane) {
    const int n  = lane & 15;
    const int kb = (lane < 16) ? 0 : 8;
    const _Float16* p = base + n * ld + kb;
    v8h lo = *(const v8h*)(p);
    v8h hi = *(const v8h*)(p + 16);
    return __builtin_shufflevector(lo, hi, 0, 1, 2, 3, 4, 5, 6, 7,
                                           8, 9, 10, 11, 12, 13, 14, 15);
}

__device__ __forceinline__ v8f wmma_f16f32(v16h a, v16h b, v8f c) {
    return __builtin_amdgcn_wmma_f32_16x16x32_f16(false, a, false, b,
                                                  (short)0, c, false, false);
}

// ---------------------------------------------------------------------------
// CDNA5 async-copy helpers
//   LDS byte address = low 32 bits of the generic (flat) shared-memory
//   address (ISA 10.2 aperture rule).
// ---------------------------------------------------------------------------
__device__ __forceinline__ void async_ld_b128(void* lds_ptr, const void* gptr) {
    unsigned int lofs = (unsigned int)(uintptr_t)lds_ptr;
    asm volatile("global_load_async_to_lds_b128 %0, %1, off"
                 :: "v"(lofs), "v"(gptr) : "memory");
}
__device__ __forceinline__ void wait_async0() {
    asm volatile("s_wait_asynccnt 0" ::: "memory");
}

// ---------------------------------------------------------------------------
// Kernel: fp32 -> fp16 conversion
// ---------------------------------------------------------------------------
__global__ void cvt_f32_f16_kernel(const float* __restrict__ src,
                                   _Float16* __restrict__ dst, int n) {
    for (int i = blockIdx.x * blockDim.x + threadIdx.x; i < n;
         i += gridDim.x * blockDim.x)
        dst[i] = (_Float16)src[i];
}

// ---------------------------------------------------------------------------
// Kernel: zero-fill (float4 wide stores) for the attn output region
// ---------------------------------------------------------------------------
__global__ void zero_f32x4_kernel(float4* __restrict__ p, long n4) {
    const float4 z = make_float4(0.f, 0.f, 0.f, 0.f);
    for (long i = blockIdx.x * (long)blockDim.x + threadIdx.x; i < n4;
         i += (long)gridDim.x * blockDim.x)
        p[i] = z;
}

// ---------------------------------------------------------------------------
// Kernel: fused QKV projection + bias + RoPE
//   Block = 4 waves sharing one 16-row A strip of x (16 x 1024 f16, 32 KB),
//   staged ONCE per block by the Tensor Data Mover (TENSOR_LOAD_TO_LDS).
//   Each wave: one 16x64 output tile (full head-dim -> RoPE pair local).
// ---------------------------------------------------------------------------
__global__ __launch_bounds__(128)
void qkv_rope_kernel(const _Float16* __restrict__ xh,
                     const _Float16* __restrict__ wq,
                     const _Float16* __restrict__ wk,
                     const _Float16* __restrict__ wv,
                     const float* __restrict__ bq,
                     const float* __restrict__ bk,
                     const float* __restrict__ bv,
                     _Float16* __restrict__ qO,
                     _Float16* __restrict__ kO,
                     _Float16* __restrict__ vO) {
    __shared__ __align__(16) _Float16 xa[16 * D_];   // 32 KB A strip

    const int lane = threadIdx.x & 31;
    const int wvid = threadIdx.x >> 5;
    const int mt  = blockIdx.x / 12;                 // 256 M tiles
    const int nt  = (blockIdx.x % 12) * 4 + wvid;    // 48 N64 tiles over 3*D
    const int M0  = mt * 16;
    const int which = nt / 16;                       // 0=q 1=k 2=v
    const int head  = nt % 16;
    const int N0d   = head * 64;

    // ---- TDM: stage A strip (16 rows x 1024 f16, row stride 1024) ----
    if (wvid == 0) {
        const unsigned long long ga =
            (unsigned long long)(uintptr_t)(xh + (size_t)M0 * D_);
        const unsigned int lofs = (unsigned int)(uintptr_t)(void*)xa;
        v4u g0;
        g0[0] = 1u;                                   // count=1, user mode
        g0[1] = lofs;                                 // lds_addr
        g0[2] = (unsigned int)(ga & 0xffffffffu);     // global_addr[31:0]
        g0[3] = (unsigned int)((ga >> 32) & 0x1ffffffu) | (2u << 30); // type=2
        v8i g1;
        g1[0] = (1 << 16);                            // data_size = 2 bytes
        g1[1] = (int)((unsigned)(D_ & 0xffff) << 16); // tensor_dim0 = 1024
        g1[2] = (int)((unsigned)(4096 & 0xffff) << 16); // tensor_dim1 = 4096
        g1[3] = (int)((unsigned)(D_ & 0xffff) << 16); // tile_dim0 = 1024
        g1[4] = 16;                                   // tile_dim1 = 16 rows
        g1[5] = D_;                                   // tensor_dim0_stride
        g1[6] = 0;
        g1[7] = 0;
        v4i zz4 = (v4i)0;
        v8i zz8 = (v8i)0;
        __builtin_amdgcn_tensor_load_to_lds(g0, g1, zz4, zz4, zz8, 0);
        __builtin_amdgcn_s_wait_tensorcnt(0);
    }
    __syncthreads();

    const _Float16* wsel = (which == 0) ? wq : (which == 1) ? wk : wv;
    const float*    bias = (which == 0) ? bq : (which == 1) ? bk : bv;

    v8f acc[4];
#pragma unroll
    for (int f = 0; f < 4; ++f) acc[f] = (v8f)0.0f;

    for (int kc = 0; kc < D_; kc += 32) {
        v16h a = load_a_frag(xa + kc, D_, lane);      // from LDS
#pragma unroll
        for (int f = 0; f < 4; ++f) {
            const _Float16* brow = wsel + (size_t)(N0d + f * 16) * D_ + kc;
            if (kc + 64 < D_)
                __builtin_prefetch(brow + (size_t)(lane & 15) * D_ + 64, 0, 1);
            acc[f] = wmma_f16f32(a, load_b_frag(brow, D_, lane), acc[f]);
        }
    }

    const int n_lane = lane & 15;
    const int moff = (lane < 16) ? 0 : 8;

    if (which == 2) {   // V: bias only
#pragma unroll
        for (int f = 0; f < 4; ++f) {
#pragma unroll
            for (int r = 0; r < 8; ++r) {
                const int row = M0 + r + moff;
                const int bb = row / S_, ss = row % S_;
                const int hd = f * 16 + n_lane;
                float val = acc[f][r] + bias[N0d + hd];
                vO[(((size_t)bb * H_ + head) * S_ + ss) * HD_ + hd] = (_Float16)val;
            }
        }
    } else {            // Q/K: bias + RoPE (pair hd <-> hd+32 = frag f <-> f+2)
        _Float16* dst = (which == 0) ? qO : kO;
        const float qscale = (which == 0) ? 0.125f : 1.0f;  // 1/sqrt(64)
#pragma unroll
        for (int f = 0; f < 2; ++f) {
#pragma unroll
            for (int r = 0; r < 8; ++r) {
                const int row = M0 + r + moff;
                const int bb = row / S_, ss = row % S_;
                const int pair = f * 16 + n_lane;            // 0..31
                float lo = acc[f][r]     + bias[N0d + pair];
                float hi = acc[f + 2][r] + bias[N0d + pair + 32];
                float theta = (float)ss * __powf(10000.0f, -(float)pair * (1.0f / 32.0f));
                float c = __cosf(theta), sn = __sinf(theta);
                float rlo = (lo * c - hi * sn) * qscale;
                float rhi = (hi * c + lo * sn) * qscale;
                size_t base = (((size_t)bb * H_ + head) * S_ + ss) * HD_;
                dst[base + pair]      = (_Float16)rlo;
                dst[base + pair + 32] = (_Float16)rhi;
            }
        }
    }
}

// ---------------------------------------------------------------------------
// Kernel: sliding-window attention
//   one workgroup (4 waves / 128 thr) per (b,h, 16-row query tile)
//   K tiles staged per-wave with GLOBAL_LOAD_ASYNC_TO_LDS_B128
// ---------------------------------------------------------------------------
__global__ __launch_bounds__(128)
void attn_window_kernel(const _Float16* __restrict__ qg,
                        const _Float16* __restrict__ kg,
                        const _Float16* __restrict__ vg,
                        float* __restrict__ attn,
                        _Float16* __restrict__ ohead) {
    const int tid = threadIdx.x;
    const int lane = tid & 31;
    const int wvid = tid >> 5;

    const int tI = blockIdx.x % (S_ / 16);
    const int bh = blockIdx.x / (S_ / 16);      // b*H + h
    const int b = bh / H_, h = bh % H_;
    const int i0 = tI * 16;

    const int jt0 = (tI >= 16) ? (tI - 16) : 0; // first key tile (WIN/16 back)
    const int jbase = jt0 * 16;
    const int ncols = i0 + 16 - jbase;          // <= 272

    __shared__ __align__(16) float    sc[16 * 272];
    __shared__ __align__(16) _Float16 pr[16 * 288];
    __shared__ __align__(16) _Float16 vT[64 * 288];     // vT[d][j_local]
    __shared__ __align__(32) _Float16 kst[4][16 * 64];  // per-wave K tile
    __shared__ float rmax[16], rsum[16];

    // ---- stage 0: zero probs pad, stage V transposed (zero OOB tail) ----
    for (int idx = tid; idx < 16 * 288; idx += 128) pr[idx] = (_Float16)0;
    for (int idx = tid; idx < 288 * 64; idx += 128) {
        const int jl = idx >> 6, d = idx & 63;
        const int jg = jbase + jl;
        _Float16 val = (_Float16)0;
        if (jg < S_) val = vg[((size_t)bh * S_ + jg) * HD_ + d];
        vT[d * 288 + jl] = val;
    }

    // ---- stage 1: scores via WMMA, mask, store to LDS ----
    const _Float16* qrow = qg + ((size_t)bh * S_ + i0) * HD_;
    v16h aq0 = load_a_frag(qrow,      HD_, lane);
    v16h aq1 = load_a_frag(qrow + 32, HD_, lane);

    const int n_lane = lane & 15;
    const int moff = (lane < 16) ? 0 : 8;

    for (int t = wvid; t * 16 < ncols; t += 4) {
        const _Float16* krow = kg + ((size_t)bh * S_ + jbase + t * 16) * HD_;
        // async-DMA the 16x64 f16 K tile (2 KB) into this wave's LDS slot:
        // 128 x 16B segments, 4 per lane
        for (int seg = lane; seg < 128; seg += 32)
            async_ld_b128(&kst[wvid][seg * 8], krow + seg * 8);
        wait_async0();   // per-wave ASYNCcnt -> no workgroup barrier needed

        v8f s8 = (v8f)0.0f;
        s8 = wmma_f16f32(aq0, load_b_frag(&kst[wvid][0],  HD_, lane), s8);
        s8 = wmma_f16f32(aq1, load_b_frag(&kst[wvid][32], HD_, lane), s8);
#pragma unroll
        for (int r = 0; r < 8; ++r) {
            const int i = i0 + r + moff;
            const int j = jbase + t * 16 + n_lane;
            float val = s8[r];
            if (!((j <= i) && (i - j < WIN_))) val = -1.0e30f;
            sc[(r + moff) * 272 + t * 16 + n_lane] = val;
        }
    }
    __syncthreads();

    // ---- stage 2: per-row max & sum (exact, window <= 272 wide) ----
    if (tid < 16) {
        const int row = tid;
        float m = -1.0e30f;
        for (int c = 0; c < ncols; ++c) {
            float s = sc[row * 272 + c];
            m = (s > m) ? s : m;
        }
        float sum = 0.0f;
        for (int c = 0; c < ncols; ++c)
            sum += __expf(sc[row * 272 + c] - m);
        rmax[row] = m;
        rsum[row] = sum;
    }
    __syncthreads();

    // ---- stage 3: probs -> LDS (f16) + windowed attn strip -> HBM ----
    const int total = 16 * ncols;
    for (int idx = tid; idx < total; idx += 128) {
        const int row = idx / ncols;
        const int col = idx - row * ncols;
        float p = __expf(sc[row * 272 + col] - rmax[row]) / rsum[row];
        pr[row * 288 + col] = (_Float16)p;
        attn[((size_t)(bh * S_ + i0 + row)) * S_ + (jbase + col)] = p;
    }
    __syncthreads();

    // ---- stage 4: O = P * V via WMMA (K padded to 288, pad probs = 0) ----
    v8f o = (v8f)0.0f;
    for (int kc = 0; kc < 288; kc += 32) {
        v16h a = load_a_frag(pr + kc, 288, lane);
        v16h bf = load_b_frag(vT + (size_t)(wvid * 16) * 288 + kc, 288, lane);
        o = wmma_f16f32(a, bf, o);
    }
#pragma unroll
    for (int r = 0; r < 8; ++r) {
        const int m = r + moff;
        const int ss = i0 + m;
        const int d = wvid * 16 + n_lane;
        ohead[((size_t)b * S_ + ss) * D_ + h * HD_ + d] = (_Float16)o[r];
    }
}

// ---------------------------------------------------------------------------
// Kernel: output projection  out = ohead @ Wo^T + bo  (fp32 result)
// ---------------------------------------------------------------------------
__global__ __launch_bounds__(128)
void oproj_kernel(const _Float16* __restrict__ ah,   // (B*S, D) f16
                  const _Float16* __restrict__ woh,  // (D, D) f16
                  const float* __restrict__ bo,
                  float* __restrict__ out) {
    const int lane = threadIdx.x & 31;
    const int wid = blockIdx.x * 4 + (threadIdx.x >> 5);  // 256*16 waves
    const int mt = wid / 16;
    const int nt = wid % 16;
    const int M0 = mt * 16;
    const int N0 = nt * 64;

    v8f acc[4];
#pragma unroll
    for (int f = 0; f < 4; ++f) acc[f] = (v8f)0.0f;

    for (int kc = 0; kc < D_; kc += 32) {
        v16h a = load_a_frag(ah + (size_t)M0 * D_ + kc, D_, lane);
#pragma unroll
        for (int f = 0; f < 4; ++f) {
            const _Float16* brow = woh + (size_t)(N0 + f * 16) * D_ + kc;
            if (kc + 64 < D_)
                __builtin_prefetch(brow + (size_t)(lane & 15) * D_ + 64, 0, 1);
            acc[f] = wmma_f16f32(a, load_b_frag(brow, D_, lane), acc[f]);
        }
    }

    const int n_lane = lane & 15;
    const int moff = (lane < 16) ? 0 : 8;
#pragma unroll
    for (int f = 0; f < 4; ++f) {
#pragma unroll
        for (int r = 0; r < 8; ++r) {
            const int row = M0 + r + moff;
            const int n = N0 + f * 16 + n_lane;
            out[(size_t)row * D_ + n] = acc[f][r] + bo[n];
        }
    }
}

// ---------------------------------------------------------------------------
// Host launcher
// ---------------------------------------------------------------------------
extern "C" void kernel_launch(void* const* d_in, const int* in_sizes, int n_in,
                              void* d_out, int out_size, void* d_ws, size_t ws_size,
                              hipStream_t stream) {
    (void)in_sizes; (void)n_in; (void)out_size; (void)ws_size;

    const float* x  = (const float*)d_in[0];
    const float* Wq = (const float*)d_in[1];
    const float* bq = (const float*)d_in[2];
    const float* Wk = (const float*)d_in[3];
    const float* bk = (const float*)d_in[4];
    const float* Wv = (const float*)d_in[5];
    const float* bv = (const float*)d_in[6];
    const float* Wo = (const float*)d_in[7];
    const float* bo = (const float*)d_in[8];

    float* out  = (float*)d_out;                               // (B,S,D)
    float* attn = out + (size_t)B_ * S_ * D_;                  // (B,H,S,S)

    // workspace layout (bytes)
    char* wsb = (char*)d_ws;
    const size_t XN = (size_t)B_ * S_ * D_;       // 4,194,304
    const size_t WN = (size_t)D_ * D_;            // 1,048,576
    _Float16* xh  = (_Float16*)(wsb);                              // 8 MB
    _Float16* wqh = (_Float16*)(wsb + XN * 2);                     // 2 MB
    _Float16* wkh = (_Float16*)(wsb + XN * 2 + WN * 2);
    _Float16* wvh = (_Float16*)(wsb + XN * 2 + WN * 4);
    _Float16* woh = (_Float16*)(wsb + XN * 2 + WN * 6);
    _Float16* qf  = (_Float16*)(wsb + XN * 2 + WN * 8);            // 8 MB each
    _Float16* kf  = qf + XN;
    _Float16* vf  = kf + XN;
    _Float16* oh  = vf + XN;

    // 1) convert fp32 -> fp16
    cvt_f32_f16_kernel<<<1024, 256, 0, stream>>>(x,  xh,  (int)XN);
    cvt_f32_f16_kernel<<<512,  256, 0, stream>>>(Wq, wqh, (int)WN);
    cvt_f32_f16_kernel<<<512,  256, 0, stream>>>(Wk, wkh, (int)WN);
    cvt_f32_f16_kernel<<<512,  256, 0, stream>>>(Wv, wvh, (int)WN);
    cvt_f32_f16_kernel<<<512,  256, 0, stream>>>(Wo, woh, (int)WN);

    // 2) zero attn region (B*H*S*S floats)
    const long attn_n4 = ((long)B_ * H_ * S_ * S_) / 4;
    zero_f32x4_kernel<<<2048, 256, 0, stream>>>((float4*)attn, attn_n4);

    // 3) QKV + RoPE: 256 M-tiles x 12 groups (4 waves of 64-wide N tiles)
    qkv_rope_kernel<<<3072, 128, 0, stream>>>(xh, wqh, wkh, wvh,
                                              bq, bk, bv, qf, kf, vf);

    // 4) attention: B*H*(S/16) = 4096 workgroups
    attn_window_kernel<<<B_ * H_ * (S_ / 16), 128, 0, stream>>>(qf, kf, vf,
                                                                attn, oh);

    // 5) output projection: 256 * 16 waves / 4 per block
    oproj_kernel<<<1024, 128, 0, stream>>>(oh, woh, bo, out);
}
